// GeometricTransformerBlock_50208167690783
// MI455X (gfx1250) — matrix operand order repaired and checked by reading, not compiled, not run on hardware
//
#include <hip/hip_runtime.h>
#include <hip/hip_bf16.h>

typedef __bf16 bf16;
typedef __attribute__((ext_vector_type(16))) __bf16 bf16x16;
typedef __attribute__((ext_vector_type(8)))  float  f32x8;
typedef __attribute__((ext_vector_type(4)))  unsigned int u32x4;
typedef __attribute__((ext_vector_type(8)))  int    i32x8;
typedef __attribute__((ext_vector_type(4)))  int    i32x4;

__device__ __forceinline__ f32x8 wmma_bf16(bf16x16 a, bf16x16 b, f32x8 c) {
  // D = A(16x32) * B(32x16) + C, f32 accumulate
  return __builtin_amdgcn_wmma_f32_16x16x32_bf16(false, a, false, b, (short)0, c, false, false);
}

// ---------------------------------------------------------------------------
// Weight prep: W[K][N] f32 (row-major, as used by x @ W) -> Wt[N][K] bf16
// so each WMMA B-fragment lane reads 16 *contiguous* K values.
// ---------------------------------------------------------------------------
__global__ void k_transpose_bf16(const float* __restrict__ W, bf16* __restrict__ Wt,
                                 int K, int N) {
  int idx = blockIdx.x * 256 + threadIdx.x;
  if (idx >= K * N) return;
  int n = idx / K;
  int k = idx - n * K;
  Wt[idx] = (bf16)W[(size_t)k * N + n];
}

// ---------------------------------------------------------------------------
// Row LayerNorm over C=256, outputs bf16 (feeds WMMA GEMMs)
// ---------------------------------------------------------------------------
__global__ void k_layernorm(const float* __restrict__ X, const float* __restrict__ g,
                            const float* __restrict__ bta, bf16* __restrict__ out, int M) {
  int row = blockIdx.x;
  int t = threadIdx.x;               // 256 threads, one element each
  __shared__ float red[256];
  float v = X[(size_t)row * 256 + t];
  red[t] = v; __syncthreads();
  for (int s = 128; s > 0; s >>= 1) { if (t < s) red[t] += red[t + s]; __syncthreads(); }
  float mean = red[0] * (1.0f / 256.0f);
  __syncthreads();
  float d = v - mean;
  red[t] = d * d; __syncthreads();
  for (int s = 128; s > 0; s >>= 1) { if (t < s) red[t] += red[t + s]; __syncthreads(); }
  float var = red[0] * (1.0f / 256.0f);
  float y = d * rsqrtf(var + 1e-5f) * g[t] + bta[t];
  out[(size_t)row * 256 + t] = (bf16)y;
}

// ---------------------------------------------------------------------------
// WMMA GEMM with TDM-staged A operand.
// Block = 8 waves, computes a 16(M) x 128(N) tile; wave w owns 16 N-columns.
// The shared A strip [16 x K] is DMA'd into LDS once per block by the
// Tensor Data Mover (TENSORcnt / s_wait_tensorcnt), then every wave reads its
// A fragments from LDS and streams B (pre-transposed bf16 weights) from L2.
// ---------------------------------------------------------------------------
__global__ void k_gemm(const bf16* __restrict__ A, const bf16* __restrict__ Wt,
                       const float* __restrict__ bias, const float* __restrict__ resid,
                       float* __restrict__ outf, bf16* __restrict__ outb,
                       int Mdim, int Ndim, int Kdim) {
  __shared__ bf16 ldsA[16 * 512];                      // up to K=512 (16 KB)
  int tid  = threadIdx.x;
  int wave = tid >> 5;
  int lane = tid & 31;
  int lm   = lane & 15;
  int half = lane >> 4;
  int nblk = Ndim >> 7;                                // N / 128
  int bm = blockIdx.x / nblk;
  int bn = blockIdx.x - bm * nblk;
  int m0 = bm << 4;
  int n0 = (bn << 7) + (wave << 4);

  // ---- TDM: stage A[m0:m0+16, 0:K] into LDS (wave 0 issues, all waves sync)
  if (wave == 0) {
    unsigned lds_base = (unsigned)(size_t)(&ldsA[0]);
    unsigned long long ga = (unsigned long long)(size_t)(A + (size_t)m0 * Kdim);
    unsigned td0 = (unsigned)Kdim;                     // tensor_dim0 (contiguous)
    unsigned td1 = 16u;                                // tensor_dim1 (rows)
    unsigned long long s0 = (unsigned long long)Kdim;  // dim0 stride (elements)
    u32x4 g0;
    g0[0] = 1u;                                        // count=1 (valid user D#)
    g0[1] = lds_base;                                  // lds_addr (bytes)
    g0[2] = (unsigned)(ga & 0xFFFFFFFFu);              // global_addr[31:0]
    g0[3] = (unsigned)((ga >> 32) & 0x1FFFFFFu)        // global_addr[56:32]
            | 0x80000000u;                             // type=2 -> bits[127:126]=10b
    i32x8 g1;
    g1[0] = (int)(1u << 16);                           // data_size=1 (2-byte elems)
    g1[1] = (int)((td0 & 0xFFFFu) << 16);              // tensor_dim0[15:0]
    g1[2] = (int)((td0 >> 16) | ((td1 & 0xFFFFu) << 16));
    g1[3] = (int)((td1 >> 16) | (td0 << 16));          // tile_dim0 = K
    g1[4] = 16;                                        // tile_dim1 = 16, tile_dim2 = 0
    g1[5] = (int)(s0 & 0xFFFFFFFFull);                 // tensor_dim0_stride[31:0]
    g1[6] = (int)((s0 >> 32) & 0xFFFFull);             // stride0[47:32], stride1=0
    g1[7] = 0;
    i32x4 z4 = {0, 0, 0, 0};
    i32x8 z8 = {0, 0, 0, 0, 0, 0, 0, 0};
    __builtin_amdgcn_tensor_load_to_lds(g0, g1, z4, z4, z8, 0);  // 6-arg form
    __builtin_amdgcn_s_wait_tensorcnt(0);
  }
  __syncthreads();

  f32x8 c = {0.f, 0.f, 0.f, 0.f, 0.f, 0.f, 0.f, 0.f};
  const bf16* arow = &ldsA[lm * Kdim];                 // A row in LDS
  const bf16* bp   = Wt + (size_t)(n0 + lm) * Kdim + half * 16;
  for (int k0 = 0; k0 < Kdim; k0 += 32) {
    if (k0 + 32 < Kdim) __builtin_prefetch(bp + k0 + 32, 0, 1);
    bf16x16 a, b;
#pragma unroll
    for (int e = 0; e < 16; ++e) {
      int kk = half * 16 + e;                          // contiguous K per lane-half
      a[e] = arow[k0 + kk];                            // ds_load from staged strip
      b[e] = bp[k0 + e];                               // 16 consecutive K per lane-half
    }
    c = wmma_bf16(a, b, c);
  }
  int col = n0 + lm;
  float bv = bias ? bias[col] : 0.f;
#pragma unroll
  for (int r = 0; r < 8; ++r) {
    int row = m0 + r + half * 8;                       // C-fragment row mapping
    float val = c[r] + bv;
    size_t o = (size_t)row * Ndim + col;
    if (resid) val += resid[o];
    if (outf)  outf[o] = val;
    if (outb)  outb[o] = (bf16)val;
  }
}

// ---------------------------------------------------------------------------
// Flash-style attention with fused SDF-bias MLP.
// One wave per (b, h, 16-query tile). Q/K/V are [B,N,H*64] bf16.
// score = (q.k)/8 - bias(|dq-dk|);  bias = relu(|d|*w1+b1) @ w2[:,h] + b2[h]
// ---------------------------------------------------------------------------
__global__ void k_attn(const bf16* __restrict__ Qb, const bf16* __restrict__ Kb,
                       const bf16* __restrict__ Vb,
                       const float* __restrict__ sdf_q, const float* __restrict__ sdf_k,
                       const float* __restrict__ w1, const float* __restrict__ b1,
                       const float* __restrict__ w2, const float* __restrict__ b2,
                       bf16* __restrict__ outb, int Bn, int Nq, int Nk) {
  __shared__ bf16 ldsP[8][16 * 32];                    // per-wave P staging
  int tid = threadIdx.x, wave = tid >> 5, lane = tid & 31;
  int lm = lane & 15, half = lane >> 4;
  int qtiles = Nq >> 4;
  int wid = blockIdx.x * 8 + wave;
  int qt = wid % qtiles;
  int h  = (wid / qtiles) & 3;
  int b  = wid / (qtiles * 4);
  int q0 = qt << 4;
  const float scale = 0.125f;                          // 64^-0.5
  const float L2E   = 1.44269504f;

  // Q fragments for hd chunks 0/1 (K-steps of 32 over hd=64)
  bf16x16 aq0, aq1;
  const bf16* qrow = Qb + (size_t)(b * Nq + q0 + lm) * 256 + h * 64;
#pragma unroll
  for (int e = 0; e < 16; ++e) {
    int kk = half * 16 + e;
    aq0[e] = qrow[kk];
    aq1[e] = qrow[32 + kk];
  }
  float dq[8];
#pragma unroll
  for (int r = 0; r < 8; ++r) dq[r] = sdf_q[b * Nq + q0 + r + half * 8];
  float w1r[16], b1r[16], w2r[16];
#pragma unroll
  for (int j = 0; j < 16; ++j) { w1r[j] = w1[j]; b1r[j] = b1[j]; w2r[j] = w2[j * 4 + h]; }
  float b2h = b2[h];

  float mrun[8], lrun[8];
#pragma unroll
  for (int r = 0; r < 8; ++r) { mrun[r] = -1e30f; lrun[r] = 0.f; }
  f32x8 o[4];
#pragma unroll
  for (int c = 0; c < 4; ++c) o[c] = (f32x8){0.f,0.f,0.f,0.f,0.f,0.f,0.f,0.f};

  for (int kb0 = 0; kb0 < Nk; kb0 += 32) {
    float p[2][8];
#pragma unroll
    for (int kk = 0; kk < 2; ++kk) {
      int kbase = kb0 + kk * 16;
      bf16x16 bk0, bk1;
      const bf16* krow = Kb + (size_t)(b * Nk + kbase + lm) * 256 + h * 64 + half * 16;
#pragma unroll
      for (int e = 0; e < 16; ++e) { bk0[e] = krow[e]; bk1[e] = krow[32 + e]; }
      f32x8 s = {0.f,0.f,0.f,0.f,0.f,0.f,0.f,0.f};
      s = wmma_bf16(aq0, bk0, s);
      s = wmma_bf16(aq1, bk1, s);
      float dk = sdf_k[b * Nk + kbase + lm];
#pragma unroll
      for (int r = 0; r < 8; ++r) {
        float t = fabsf(dq[r] - dk);
        float acc = b2h;
#pragma unroll
        for (int j = 0; j < 16; ++j)
          acc += fmaxf(t * w1r[j] + b1r[j], 0.f) * w2r[j];
        p[kk][r] = s[r] * scale - acc;                 // softmax(score - bias)
      }
    }
    // Online softmax update; row data for a given row lives in one lane-half,
    // so xor-shuffles with mask<16 reduce over the 16 key columns.
#pragma unroll
    for (int r = 0; r < 8; ++r) {
      float mx = fmaxf(p[0][r], p[1][r]);
      for (int d = 8; d >= 1; d >>= 1) mx = fmaxf(mx, __shfl_xor(mx, d));
      float mnew = fmaxf(mrun[r], mx);
      float e0 = exp2f((p[0][r] - mnew) * L2E);
      float e1 = exp2f((p[1][r] - mnew) * L2E);
      float alpha = exp2f((mrun[r] - mnew) * L2E);
      float rs = e0 + e1;
      for (int d = 8; d >= 1; d >>= 1) rs += __shfl_xor(rs, d);
      lrun[r] = lrun[r] * alpha + rs;
      mrun[r] = mnew;
#pragma unroll
      for (int c = 0; c < 4; ++c) o[c][r] *= alpha;
      int row = r + half * 8;
      ldsP[wave][row * 32 + lm]      = (bf16)e0;       // C-layout -> LDS
      ldsP[wave][row * 32 + 16 + lm] = (bf16)e1;
    }
    // Re-load P in A-fragment layout (LDS ops are in-order per wave)
    bf16x16 ap;
#pragma unroll
    for (int e = 0; e < 16; ++e) {
      int kc = half * 16 + e;
      ap[e] = ldsP[wave][lm * 32 + kc];
    }
    // O += P(16x32) @ V(32x64), per 16-wide hd chunk
#pragma unroll
    for (int c = 0; c < 4; ++c) {
      bf16x16 bv;
#pragma unroll
      for (int e = 0; e < 16; ++e) {
        int key = kb0 + half * 16 + e;
        bv[e] = Vb[(size_t)(b * Nk + key) * 256 + h * 64 + c * 16 + lm];
      }
      o[c] = wmma_bf16(ap, bv, o[c]);
    }
  }
#pragma unroll
  for (int r = 0; r < 8; ++r) {
    float inv = 1.f / lrun[r];
    int row = q0 + r + half * 8;
#pragma unroll
    for (int c = 0; c < 4; ++c)
      outb[(size_t)(b * Nq + row) * 256 + h * 64 + c * 16 + lm] = (bf16)(o[c][r] * inv);
  }
}

// ---------------------------------------------------------------------------
// Geometric MLP activation: y[M,1024] -> g[M,512]
// xL=y[:, :512], xR=y[:, 512:]; H=xL*xR+xL+xR; out = silu(H)*H
// ---------------------------------------------------------------------------
__global__ void k_geo_act(const float* __restrict__ y, bf16* __restrict__ g, int M) {
  int idx = blockIdx.x * 256 + threadIdx.x;
  if (idx >= M * 512) return;
  int row = idx >> 9, i = idx & 511;
  float xL = y[(size_t)row * 1024 + i];
  float xR = y[(size_t)row * 1024 + 512 + i];
  float H = xL * xR + xL + xR;
  float s = H / (1.f + __expf(-H));                    // silu(H)
  g[idx] = (bf16)(s * H);
}

// ---------------------------------------------------------------------------
extern "C" void kernel_launch(void* const* d_in, const int* in_sizes, int n_in,
                              void* d_out, int out_size, void* d_ws, size_t ws_size,
                              hipStream_t stream) {
  (void)in_sizes; (void)n_in; (void)out_size; (void)ws_size;
  const int Bn = 2, Nn = 1536, M = Bn * Nn;

  const float* x     = (const float*)d_in[0];
  const float* src   = (const float*)d_in[1];
  const float* xsdf  = (const float*)d_in[2];
  const float* ssdf  = (const float*)d_in[3];
  const float* sa_wq = (const float*)d_in[4];  const float* sa_bq = (const float*)d_in[5];
  const float* sa_wk = (const float*)d_in[6];  const float* sa_bk = (const float*)d_in[7];
  const float* sa_wv = (const float*)d_in[8];  const float* sa_bv = (const float*)d_in[9];
  const float* sa_wp = (const float*)d_in[10]; const float* sa_bp = (const float*)d_in[11];
  const float* sa_w1 = (const float*)d_in[12]; const float* sa_b1 = (const float*)d_in[13];
  const float* sa_w2 = (const float*)d_in[14]; const float* sa_b2 = (const float*)d_in[15];
  const float* ca_wq = (const float*)d_in[16]; const float* ca_bq = (const float*)d_in[17];
  const float* ca_wk = (const float*)d_in[18]; const float* ca_bk = (const float*)d_in[19];
  const float* ca_wv = (const float*)d_in[20]; const float* ca_bv = (const float*)d_in[21];
  const float* ca_wp = (const float*)d_in[22]; const float* ca_bp = (const float*)d_in[23];
  const float* ca_w1 = (const float*)d_in[24]; const float* ca_b1 = (const float*)d_in[25];
  const float* ca_w2 = (const float*)d_in[26]; const float* ca_b2 = (const float*)d_in[27];
  const float* w_fc1 = (const float*)d_in[28]; const float* b_fc1 = (const float*)d_in[29];
  const float* w_fc2 = (const float*)d_in[30]; const float* b_fc2 = (const float*)d_in[31];
  const float* ln1g  = (const float*)d_in[32]; const float* ln1b  = (const float*)d_in[33];
  const float* ln2g  = (const float*)d_in[34]; const float* ln2b  = (const float*)d_in[35];
  const float* ln3g  = (const float*)d_in[36]; const float* ln3b  = (const float*)d_in[37];

  // Workspace carving
  char* ws = (char*)d_ws;
  auto carve = [&](size_t bytes) -> char* {
    char* p = ws; ws += (bytes + 255) & ~(size_t)255; return p;
  };
  const size_t WSZ = 256 * 256 * sizeof(bf16);         // 256x256 bf16 weight
  const size_t ACT = (size_t)M * 256 * sizeof(bf16);   // bf16 activation [M,256]
  bf16* t_sa_wq = (bf16*)carve(WSZ); bf16* t_sa_wk = (bf16*)carve(WSZ);
  bf16* t_sa_wv = (bf16*)carve(WSZ); bf16* t_sa_wp = (bf16*)carve(WSZ);
  bf16* t_ca_wq = (bf16*)carve(WSZ); bf16* t_ca_wk = (bf16*)carve(WSZ);
  bf16* t_ca_wv = (bf16*)carve(WSZ); bf16* t_ca_wp = (bf16*)carve(WSZ);
  bf16* t_fc1   = (bf16*)carve((size_t)1024 * 256 * sizeof(bf16)); // [1024][256]
  bf16* t_fc2   = (bf16*)carve((size_t)256 * 512 * sizeof(bf16));  // [256][512]
  bf16* n1b   = (bf16*)carve(ACT);
  bf16* qb    = (bf16*)carve(ACT);
  bf16* kb    = (bf16*)carve(ACT);
  bf16* vb    = (bf16*)carve(ACT);
  bf16* attnb = (bf16*)carve(ACT);
  bf16* nqb   = (bf16*)carve(ACT);
  bf16* nkb   = (bf16*)carve(ACT);
  bf16* n3b   = (bf16*)carve(ACT);
  bf16* gb    = (bf16*)carve((size_t)M * 512 * sizeof(bf16));
  float* x1   = (float*)carve((size_t)M * 256 * sizeof(float));
  float* x2   = (float*)carve((size_t)M * 256 * sizeof(float));
  float* y1   = (float*)carve((size_t)M * 1024 * sizeof(float));

  // --- weight prep (bf16 transpose) ---
  auto tpose = [&](const float* W, bf16* Wt, int K, int N) {
    k_transpose_bf16<<<(K * N + 255) / 256, 256, 0, stream>>>(W, Wt, K, N);
  };
  tpose(sa_wq, t_sa_wq, 256, 256); tpose(sa_wk, t_sa_wk, 256, 256);
  tpose(sa_wv, t_sa_wv, 256, 256); tpose(sa_wp, t_sa_wp, 256, 256);
  tpose(ca_wq, t_ca_wq, 256, 256); tpose(ca_wk, t_ca_wk, 256, 256);
  tpose(ca_wv, t_ca_wv, 256, 256); tpose(ca_wp, t_ca_wp, 256, 256);
  tpose(w_fc1, t_fc1, 256, 1024);  tpose(w_fc2, t_fc2, 512, 256);

  auto gemm = [&](const bf16* A, const bf16* Wt, const float* bias, const float* resid,
                  float* outf, bf16* outb, int Md, int Nd, int Kd) {
    int blocks = (Md >> 4) * (Nd >> 7);                // 16-row x 128-col tiles
    k_gemm<<<blocks, 256, 0, stream>>>(A, Wt, bias, resid, outf, outb, Md, Nd, Kd);
  };
  const int attn_blocks = (Bn * 4 * (Nn >> 4)) / 8;    // 96

  // --- self attention ---
  k_layernorm<<<M, 256, 0, stream>>>(x, ln1g, ln1b, n1b, M);
  gemm(n1b, t_sa_wq, sa_bq, nullptr, nullptr, qb, M, 256, 256);
  gemm(n1b, t_sa_wk, sa_bk, nullptr, nullptr, kb, M, 256, 256);
  gemm(n1b, t_sa_wv, sa_bv, nullptr, nullptr, vb, M, 256, 256);
  k_attn<<<attn_blocks, 256, 0, stream>>>(qb, kb, vb, xsdf, xsdf,
                                          sa_w1, sa_b1, sa_w2, sa_b2, attnb, Bn, Nn, Nn);
  gemm(attnb, t_sa_wp, sa_bp, x, x1, nullptr, M, 256, 256);   // x1 = x + attn

  // --- cross attention ---
  k_layernorm<<<M, 256, 0, stream>>>(x1, ln2g, ln2b, nqb, M);
  k_layernorm<<<M, 256, 0, stream>>>(src, ln2g, ln2b, nkb, M);
  gemm(nqb, t_ca_wq, ca_bq, nullptr, nullptr, qb, M, 256, 256);
  gemm(nkb, t_ca_wk, ca_bk, nullptr, nullptr, kb, M, 256, 256);
  gemm(nkb, t_ca_wv, ca_bv, nullptr, nullptr, vb, M, 256, 256);
  k_attn<<<attn_blocks, 256, 0, stream>>>(qb, kb, vb, xsdf, ssdf,
                                          ca_w1, ca_b1, ca_w2, ca_b2, attnb, Bn, Nn, Nn);
  gemm(attnb, t_ca_wp, ca_bp, x1, x2, nullptr, M, 256, 256);  // x2 = x1 + attn

  // --- geometric MLP ---
  k_layernorm<<<M, 256, 0, stream>>>(x2, ln3g, ln3b, n3b, M);
  gemm(n3b, t_fc1, b_fc1, nullptr, y1, nullptr, M, 1024, 256);
  k_geo_act<<<(M * 512 + 255) / 256, 256, 0, stream>>>(y1, gb, M);
  gemm(gb, t_fc2, b_fc2, x2, (float*)d_out, nullptr, M, 256, 512);  // out = x2 + mlp
}
